// Autoformer_47949014892849
// MI455X (gfx1250) — compile-verified, matrix operand order checked by reading, not run on hardware
//
#include <hip/hip_runtime.h>

// ---------------------------------------------------------------------------
// Autoformer encoder forward for MI455X (gfx1250, wave32, WMMA).
// All large GEMMs (QKV/O/FFN and the Gram-matrix autocorrelation) run on
// v_wmma_f32_16x16x32_bf16 with fp32 accumulation.
// ---------------------------------------------------------------------------

typedef unsigned short u16;
typedef __attribute__((ext_vector_type(16))) __bf16 v16bf;
typedef __attribute__((ext_vector_type(8)))  float  v8f;

#define BB      32
#define LL      1024
#define DM      512
#define TOPK    20
#define MAVG    25
#define BLD     ((size_t)BB * LL * DM)   // 16,777,216 elements

__device__ __forceinline__ u16 f2bf(float f) {      // RNE f32 -> bf16
  unsigned u = __float_as_uint(f);
  unsigned r = u + 0x7FFFu + ((u >> 16) & 1u);
  return (u16)(r >> 16);
}
__device__ __forceinline__ float bf2f(u16 h) { return __uint_as_float(((unsigned)h) << 16); }

union Frag16 { uint4 q[2]; v16bf v; };

// ---------------------------------------------------------------------------
// Generic elementwise helpers
// ---------------------------------------------------------------------------
__global__ void k_cvt_f32_bf16(const float* __restrict__ s, u16* __restrict__ d, int n) {
  for (int i = blockIdx.x * 256 + threadIdx.x; i < n; i += gridDim.x * 256) d[i] = f2bf(s[i]);
}
__global__ void k_zero_f32(float* __restrict__ p, int n) {
  for (int i = blockIdx.x * 256 + threadIdx.x; i < n; i += gridDim.x * 256) p[i] = 0.0f;
}
__global__ void k_add(const float* __restrict__ a, const float* __restrict__ b,
                      float* __restrict__ c, int n) {
  for (int i = blockIdx.x * 256 + threadIdx.x; i < n; i += gridDim.x * 256) c[i] = a[i] + b[i];
}

// ---------------------------------------------------------------------------
// Token embedding: circular conv1d k=3, [B,L,16] x [512,16,3] -> [B,L,512]
// ---------------------------------------------------------------------------
__global__ __launch_bounds__(256) void k_embed(const float* __restrict__ xe,
                                               const float* __restrict__ w,
                                               float* __restrict__ X) {
  const int l = blockIdx.x, b = blockIdx.y, tid = threadIdx.x;
  __shared__ float xin[48];
  if (tid < 48) {
    int j = tid / 16, c = tid % 16;
    int ls = (l + j - 1 + LL) & (LL - 1);              // circular pad
    xin[tid] = xe[((size_t)b * LL + ls) * 16 + c];
  }
  __syncthreads();
  for (int d = tid; d < DM; d += 256) {
    float s = 0.0f;
#pragma unroll
    for (int j = 0; j < 3; ++j)
#pragma unroll
      for (int c = 0; c < 16; ++c)
        s += xin[j * 16 + c] * w[((size_t)d * 16 + c) * 3 + j];
    X[((size_t)b * LL + l) * DM + d] = s;
  }
}

// ---------------------------------------------------------------------------
// WMMA bf16 GEMM:  D[m,n] = sum_k A[m,k] * W[n,k]  (+bias) (gelu) -> f32|bf16
// Block tile 128x128, 8 waves (2x4), each wave 64x32 (4x2 wmma accums), K-step 32.
// MODE bits: 1=bias, 2=gelu(exact erf), 4=store bf16 (else f32)
// ---------------------------------------------------------------------------
template <int MODE>
__global__ __launch_bounds__(256) void k_gemm_bf16(const u16* __restrict__ A,
                                                   const u16* __restrict__ W,
                                                   const float* __restrict__ bias,
                                                   float* __restrict__ outF,
                                                   u16* __restrict__ outB,
                                                   int M, int N, int K) {
  __shared__ __align__(16) u16 As[128 * 32];
  __shared__ __align__(16) u16 Bs[128 * 32];
  const int tid = threadIdx.x, lane = tid & 31, wave = tid >> 5;
  const int wm = wave >> 2, wn = wave & 3;             // 2 x 4 wave grid
  const int m0 = blockIdx.x * 128, n0 = blockIdx.y * 128;
  const int lrow = tid >> 1, lcol = (tid & 1) * 16;

  const v8f vz = {0.f, 0.f, 0.f, 0.f, 0.f, 0.f, 0.f, 0.f};
  v8f acc[4][2];
#pragma unroll
  for (int mt = 0; mt < 4; ++mt)
#pragma unroll
    for (int nt = 0; nt < 2; ++nt) acc[mt][nt] = vz;

  const int koff = (lane >> 4) * 8;                    // A/B fragment k-offset per half-wave
  for (int kk = 0; kk < K; kk += 32) {
    const u16* ga = A + (size_t)(m0 + lrow) * K + kk + lcol;
    const u16* gb = W + (size_t)(n0 + lrow) * K + kk + lcol;
    *(uint4*)&As[lrow * 32 + lcol]     = *(const uint4*)ga;
    *(uint4*)&As[lrow * 32 + lcol + 8] = *(const uint4*)(ga + 8);
    *(uint4*)&Bs[lrow * 32 + lcol]     = *(const uint4*)gb;
    *(uint4*)&Bs[lrow * 32 + lcol + 8] = *(const uint4*)(gb + 8);
    if (kk + 32 < K) {                                  // global_prefetch_b8 next K-slab
      __builtin_prefetch(ga + 32, 0, 1);
      __builtin_prefetch(gb + 32, 0, 1);
    }
    __syncthreads();

    Frag16 a[4], b[2];
#pragma unroll
    for (int mt = 0; mt < 4; ++mt) {
      int mr = wm * 64 + mt * 16 + (lane & 15);
      a[mt].q[0] = *(const uint4*)&As[mr * 32 + koff];
      a[mt].q[1] = *(const uint4*)&As[mr * 32 + koff + 16];
    }
#pragma unroll
    for (int nt = 0; nt < 2; ++nt) {
      int nr = wn * 32 + nt * 16 + (lane & 15);
      b[nt].q[0] = *(const uint4*)&Bs[nr * 32 + koff];
      b[nt].q[1] = *(const uint4*)&Bs[nr * 32 + koff + 16];
    }
#pragma unroll
    for (int mt = 0; mt < 4; ++mt)
#pragma unroll
      for (int nt = 0; nt < 2; ++nt)
        acc[mt][nt] = __builtin_amdgcn_wmma_f32_16x16x32_bf16(
            false, a[mt].v, false, b[nt].v, (short)0, acc[mt][nt], false, false);
    __syncthreads();
  }

#pragma unroll
  for (int mt = 0; mt < 4; ++mt) {
    const int mbase = m0 + wm * 64 + mt * 16 + ((lane >> 4) << 3);
#pragma unroll
    for (int nt = 0; nt < 2; ++nt) {
      const int n = n0 + wn * 32 + nt * 16 + (lane & 15);
      const float bv = (MODE & 1) ? bias[n] : 0.0f;
#pragma unroll
      for (int r = 0; r < 8; ++r) {
        float v = acc[mt][nt][r] + bv;
        if (MODE & 2) v = 0.5f * v * (1.0f + erff(v * 0.70710678118654752f));
        size_t o = (size_t)(mbase + r) * N + n;
        if (MODE & 4) outB[o] = f2bf(v);
        else          outF[o] = v;
      }
    }
  }
}

// ---------------------------------------------------------------------------
// Autocorrelation as Gram-matrix diagonals:
//   MV[b,tau] += sum over tile of Q[b,i,:].K[b,j,:] with tau=(i-j) mod L
// Same WMMA main loop; epilogue reduces along diagonals via LDS float atomics.
// ---------------------------------------------------------------------------
__global__ __launch_bounds__(256) void k_corr_diag(const u16* __restrict__ Qb,
                                                   const u16* __restrict__ Kb,
                                                   float* __restrict__ MV) {
  __shared__ __align__(16) u16 As[128 * 32];
  __shared__ __align__(16) u16 Bs[128 * 32];
  __shared__ float bins[LL];
  const int tid = threadIdx.x, lane = tid & 31, wave = tid >> 5;
  const int wm = wave >> 2, wn = wave & 3;
  const int b = blockIdx.z;
  const int m0 = blockIdx.x * 128, n0 = blockIdx.y * 128;
  const u16* A = Qb + (size_t)b * LL * DM;
  const u16* W = Kb + (size_t)b * LL * DM;
  const int lrow = tid >> 1, lcol = (tid & 1) * 16;

  for (int i = tid; i < LL; i += 256) bins[i] = 0.0f;

  const v8f vz = {0.f, 0.f, 0.f, 0.f, 0.f, 0.f, 0.f, 0.f};
  v8f acc[4][2];
#pragma unroll
  for (int mt = 0; mt < 4; ++mt)
#pragma unroll
    for (int nt = 0; nt < 2; ++nt) acc[mt][nt] = vz;

  const int koff = (lane >> 4) * 8;
  for (int kk = 0; kk < DM; kk += 32) {
    const u16* ga = A + (size_t)(m0 + lrow) * DM + kk + lcol;
    const u16* gb = W + (size_t)(n0 + lrow) * DM + kk + lcol;
    *(uint4*)&As[lrow * 32 + lcol]     = *(const uint4*)ga;
    *(uint4*)&As[lrow * 32 + lcol + 8] = *(const uint4*)(ga + 8);
    *(uint4*)&Bs[lrow * 32 + lcol]     = *(const uint4*)gb;
    *(uint4*)&Bs[lrow * 32 + lcol + 8] = *(const uint4*)(gb + 8);
    __syncthreads();
    Frag16 a[4], bfr[2];
#pragma unroll
    for (int mt = 0; mt < 4; ++mt) {
      int mr = wm * 64 + mt * 16 + (lane & 15);
      a[mt].q[0] = *(const uint4*)&As[mr * 32 + koff];
      a[mt].q[1] = *(const uint4*)&As[mr * 32 + koff + 16];
    }
#pragma unroll
    for (int nt = 0; nt < 2; ++nt) {
      int nr = wn * 32 + nt * 16 + (lane & 15);
      bfr[nt].q[0] = *(const uint4*)&Bs[nr * 32 + koff];
      bfr[nt].q[1] = *(const uint4*)&Bs[nr * 32 + koff + 16];
    }
#pragma unroll
    for (int mt = 0; mt < 4; ++mt)
#pragma unroll
      for (int nt = 0; nt < 2; ++nt)
        acc[mt][nt] = __builtin_amdgcn_wmma_f32_16x16x32_bf16(
            false, a[mt].v, false, bfr[nt].v, (short)0, acc[mt][nt], false, false);
    __syncthreads();
  }

  // diagonal reduce:  G[i,j] -> bins[(i-j) mod L]  (ds_add_f32)
#pragma unroll
  for (int mt = 0; mt < 4; ++mt) {
    const int ibase = m0 + wm * 64 + mt * 16 + ((lane >> 4) << 3);
#pragma unroll
    for (int nt = 0; nt < 2; ++nt) {
      const int j = n0 + wn * 32 + nt * 16 + (lane & 15);
#pragma unroll
      for (int r = 0; r < 8; ++r)
        atomicAdd(&bins[(ibase + r - j) & (LL - 1)], acc[mt][nt][r]);
    }
  }
  __syncthreads();
  for (int t = tid; t < LL; t += 256) {
    float v = bins[t];
    if (v != 0.0f) atomicAdd(&MV[b * LL + t], v);
  }
}

// ---------------------------------------------------------------------------
// Top-20 delays + softmax per batch (one workgroup per batch)
// ---------------------------------------------------------------------------
__global__ __launch_bounds__(256) void k_topk(const float* __restrict__ MV,
                                              float* __restrict__ P, int* __restrict__ DL) {
  const int b = blockIdx.x, tid = threadIdx.x;
  __shared__ float vals[LL];
  __shared__ float rv[256];
  __shared__ int   ri[256];
  __shared__ float topv[TOPK];
  __shared__ int   topi[TOPK];
  for (int i = tid; i < LL; i += 256) vals[i] = MV[b * LL + i] * (1.0f / (float)DM);
  __syncthreads();
  for (int it = 0; it < TOPK; ++it) {
    float best = -3.0e38f; int bi = 0;
    for (int i = tid; i < LL; i += 256) { float v = vals[i]; if (v > best) { best = v; bi = i; } }
    rv[tid] = best; ri[tid] = bi; __syncthreads();
    for (int s = 128; s > 0; s >>= 1) {
      if (tid < s && rv[tid + s] > rv[tid]) { rv[tid] = rv[tid + s]; ri[tid] = ri[tid + s]; }
      __syncthreads();
    }
    if (tid == 0) { topv[it] = rv[0]; topi[it] = ri[0]; vals[ri[0]] = -3.0e38f; }
    __syncthreads();
  }
  if (tid == 0) {
    float mx = topv[0];
#pragma unroll
    for (int i = 1; i < TOPK; ++i) mx = fmaxf(mx, topv[i]);
    float e[TOPK], s = 0.0f;
#pragma unroll
    for (int i = 0; i < TOPK; ++i) { e[i] = __expf(topv[i] - mx); s += e[i]; }
    float inv = 1.0f / s;
#pragma unroll
    for (int i = 0; i < TOPK; ++i) { P[b * TOPK + i] = e[i] * inv; DL[b * TOPK + i] = topi[i]; }
  }
}

// ---------------------------------------------------------------------------
// Time-delay aggregation: out[b,t,d] = sum_i p_i * V[b,(t+delay_i)%L,d]
// ---------------------------------------------------------------------------
__global__ __launch_bounds__(256) void k_agg(const u16* __restrict__ V,
                                             const float* __restrict__ P,
                                             const int* __restrict__ DL,
                                             u16* __restrict__ out) {
  const int t = blockIdx.x, b = blockIdx.y, tid = threadIdx.x;
  __shared__ float p[TOPK];
  __shared__ int   dl[TOPK];
  if (tid < TOPK) { p[tid] = P[b * TOPK + tid]; dl[tid] = DL[b * TOPK + tid]; }
  __syncthreads();
  for (int d = tid; d < DM; d += 256) {
    float s = 0.0f;
#pragma unroll
    for (int i = 0; i < TOPK; ++i) {
      int tt = (t + dl[i]) & (LL - 1);
      s += p[i] * bf2f(V[((size_t)b * LL + tt) * DM + d]);
    }
    out[((size_t)b * LL + t) * DM + d] = f2bf(s);
  }
}

// ---------------------------------------------------------------------------
// Series decomp: O = S - movmean(S, 25) with replicate padding (sliding window)
// One thread per (b,d); coalesced across d at each timestep.
// ---------------------------------------------------------------------------
__global__ __launch_bounds__(256) void k_decomp(const float* __restrict__ S,
                                                float* __restrict__ O) {
  const int idx = blockIdx.x * 256 + threadIdx.x;     // over B*D
  if (idx >= BB * DM) return;
  const int b = idx / DM, d = idx % DM;
  const float* s = S + (size_t)b * LL * DM + d;
  float* o = O + (size_t)b * LL * DM + d;
  const int pad = (MAVG - 1) / 2;                     // 12
  float win = 0.0f;
#pragma unroll
  for (int j = -pad; j <= pad; ++j) {
    int tt = j < 0 ? 0 : j;
    win += s[(size_t)tt * DM];
  }
  for (int t = 0; t < LL; ++t) {
    o[(size_t)t * DM] = s[(size_t)t * DM] - win * (1.0f / (float)MAVG);
    int tin = t + pad + 1;  if (tin > LL - 1) tin = LL - 1;
    int tout = t - pad;     if (tout < 0)     tout = 0;
    win += s[(size_t)tin * DM] - s[(size_t)tout * DM];
  }
}

// ---------------------------------------------------------------------------
// my_Layernorm (per-token), then column mean over time, then final mean
// ---------------------------------------------------------------------------
__global__ __launch_bounds__(256) void k_ln(const float* __restrict__ X,
                                            const float* __restrict__ g,
                                            const float* __restrict__ be,
                                            float* __restrict__ O) {
  const int bt = blockIdx.x, tid = threadIdx.x;
  const float* x = X + (size_t)bt * DM;
  __shared__ float ssum[256], ssq[256];
  float v0 = x[tid], v1 = x[tid + 256];
  ssum[tid] = v0 + v1; ssq[tid] = v0 * v0 + v1 * v1;
  __syncthreads();
  for (int s = 128; s > 0; s >>= 1) {
    if (tid < s) { ssum[tid] += ssum[tid + s]; ssq[tid] += ssq[tid + s]; }
    __syncthreads();
  }
  const float mu = ssum[0] * (1.0f / DM);
  const float var = ssq[0] * (1.0f / DM) - mu * mu;
  const float rstd = rsqrtf(var + 1e-5f);
  float* o = O + (size_t)bt * DM;
  o[tid]       = (v0 - mu) * rstd * g[tid]       + be[tid];
  o[tid + 256] = (v1 - mu) * rstd * g[tid + 256] + be[tid + 256];
}

__global__ __launch_bounds__(256) void k_colmean(const float* __restrict__ X,
                                                 float* __restrict__ CM) {
  const int idx = blockIdx.x * 256 + threadIdx.x;     // over B*D
  if (idx >= BB * DM) return;
  const int b = idx / DM, d = idx % DM;
  const float* x = X + (size_t)b * LL * DM + d;
  float s = 0.0f;
  for (int t = 0; t < LL; ++t) s += x[(size_t)t * DM];
  CM[idx] = s * (1.0f / (float)LL);
}

__global__ __launch_bounds__(256) void k_final(const float* __restrict__ X,
                                               const float* __restrict__ CM,
                                               float* __restrict__ out) {
  const int idx = blockIdx.x * 256 + threadIdx.x;     // over B*D
  if (idx >= BB * DM) return;
  const int b = idx / DM, d = idx % DM;
  const float* x = X + (size_t)b * LL * DM + d;
  const float cm = CM[idx];
  float s = 0.0f;
  for (int t = 0; t < LL; ++t) s += x[(size_t)t * DM] - cm;
  out[idx] = s * (1.0f / (float)LL);
}

// ---------------------------------------------------------------------------
// Host orchestration
// ---------------------------------------------------------------------------
extern "C" void kernel_launch(void* const* d_in, const int* in_sizes, int n_in,
                              void* d_out, int out_size, void* d_ws, size_t ws_size,
                              hipStream_t stream) {
  (void)in_sizes; (void)n_in; (void)out_size; (void)ws_size;
  const float* x_enc = (const float*)d_in[0];
  const float* emb_w = (const float*)d_in[1];
  const float* Wm[6] = {(const float*)d_in[2], (const float*)d_in[4], (const float*)d_in[6],
                        (const float*)d_in[8], (const float*)d_in[10], (const float*)d_in[11]};
  const float* bq = (const float*)d_in[3];
  const float* bk = (const float*)d_in[5];
  const float* bv = (const float*)d_in[7];
  const float* bo = (const float*)d_in[9];
  const float* ng = (const float*)d_in[12];
  const float* nb = (const float*)d_in[13];
  float* out = (float*)d_out;

  char* w = (char*)d_ws;
  float* X  = (float*)w;  w += BLD * 4;
  float* T0 = (float*)w;  w += BLD * 4;
  float* T1 = (float*)w;  w += BLD * 4;
  u16* Xb = (u16*)w;      w += BLD * 2;
  u16* Qb = (u16*)w;      w += BLD * 2;
  u16* Kb = (u16*)w;      w += BLD * 2;
  u16* Vb = (u16*)w;      w += BLD * 2;
  const size_t WELEM = (size_t)2 * DM * DM;            // per stacked weight tensor
  u16* WB[6];
  for (int i = 0; i < 6; ++i) { WB[i] = (u16*)w; w += WELEM * 2; }
  float* MV = (float*)w;  w += (size_t)BB * LL * 4;
  float* P  = (float*)w;  w += (size_t)BB * TOPK * 4;
  int*   DL = (int*)w;    w += (size_t)BB * TOPK * 4;
  float* CM = (float*)w;  w += (size_t)BB * DM * 4;

  const int M = BB * LL;                               // 32768
  const dim3 gemm_grid(M / 128, DM / 128);             // 256 x 4
  const dim3 corr_grid(LL / 128, LL / 128, BB);        // 8 x 8 x 32
  const dim3 bl_grid(LL, BB);                          // 1024 x 32
  const int bd_blocks = (BB * DM + 255) / 256;         // 64

  // 1. embedding + weight conversion to bf16
  k_embed<<<bl_grid, 256, 0, stream>>>(x_enc, emb_w, X);
  for (int i = 0; i < 6; ++i)
    k_cvt_f32_bf16<<<512, 256, 0, stream>>>(Wm[i], WB[i], (int)WELEM);

  for (int layer = 0; layer < 2; ++layer) {
    const size_t woff = (size_t)layer * DM * DM;
    const size_t boff = (size_t)layer * DM;

    k_cvt_f32_bf16<<<2048, 256, 0, stream>>>(X, Xb, (int)BLD);
    k_zero_f32<<<128, 256, 0, stream>>>(MV, BB * LL);

    // Q, K, V projections (bias + bf16 store)
    k_gemm_bf16<5><<<gemm_grid, 256, 0, stream>>>(Xb, WB[0] + woff, bq + boff, nullptr, Qb, M, DM, DM);
    k_gemm_bf16<5><<<gemm_grid, 256, 0, stream>>>(Xb, WB[1] + woff, bk + boff, nullptr, Kb, M, DM, DM);
    k_gemm_bf16<5><<<gemm_grid, 256, 0, stream>>>(Xb, WB[2] + woff, bv + boff, nullptr, Vb, M, DM, DM);

    // autocorrelation (Gram diagonals), top-k + softmax, delay aggregation
    k_corr_diag<<<corr_grid, 256, 0, stream>>>(Qb, Kb, MV);
    k_topk<<<BB, 256, 0, stream>>>(MV, P, DL);
    k_agg<<<bl_grid, 256, 0, stream>>>(Vb, P, DL, Xb);     // Xb <- agg (bf16)

    // output projection + residual + decomp1
    k_gemm_bf16<1><<<gemm_grid, 256, 0, stream>>>(Xb, WB[3] + woff, bo + boff, T0, nullptr, M, DM, DM);
    k_add<<<2048, 256, 0, stream>>>(X, T0, T1, (int)BLD);
    k_decomp<<<bd_blocks, 256, 0, stream>>>(T1, X);

    // FFN: gelu(X @ W1^T) @ W2^T + residual + decomp2
    k_cvt_f32_bf16<<<2048, 256, 0, stream>>>(X, Xb, (int)BLD);
    k_gemm_bf16<6><<<gemm_grid, 256, 0, stream>>>(Xb, WB[4] + woff, nullptr, nullptr, Qb, M, DM, DM);
    k_gemm_bf16<0><<<gemm_grid, 256, 0, stream>>>(Qb, WB[5] + woff, nullptr, T0, nullptr, M, DM, DM);
    k_add<<<2048, 256, 0, stream>>>(X, T0, T1, (int)BLD);
    k_decomp<<<bd_blocks, 256, 0, stream>>>(T1, X);
  }

  // final my_Layernorm + per-series time-demeaning + time-mean
  k_ln<<<M, 256, 0, stream>>>(X, ng, nb, T0);
  k_colmean<<<bd_blocks, 256, 0, stream>>>(T0, CM);
  k_final<<<bd_blocks, 256, 0, stream>>>(T0, CM, out);
}
